// MLPMoE_21079699488837
// MI455X (gfx1250) — matrix operand order repaired and checked by reading, not compiled, not run on hardware
//
#include <hip/hip_runtime.h>
#include <math.h>

// ---------------- problem constants ----------------
#define BATCH 8
#define NTOK  576
#define BN    4608          // BATCH*NTOK tokens
#define MDIM  1024
#define CDIM  2048
#define NEXP  8
#define GRAMD (NTOK*CDIM)   // 1179648, flattened per-batch dim for diff loss

// LDS tile geometry: 128 rows x 32 bf16 (64B data) padded to 80B rows -> bank-conflict-free
#define LDS_ROW   80
#define LDS_TILE  (128 * LDS_ROW)     // 10240 B per matrix tile
#define LDS_STAGE (2 * LDS_TILE)      // 20480 B per double-buffer stage (A + B)

typedef __attribute__((ext_vector_type(16))) __bf16 v16bf;
typedef __attribute__((ext_vector_type(8)))  float  v8f;
typedef __attribute__((ext_vector_type(4)))  unsigned int v4u;
typedef __attribute__((ext_vector_type(4)))  int    v4i;
typedef __attribute__((ext_vector_type(8)))  int    v8i;

// ---- TDM availability / arity detection ----
#if defined(__has_builtin)
#if __has_builtin(__builtin_amdgcn_tensor_load_to_lds)
#define HAVE_TDM 1
#endif
#endif

#ifdef HAVE_TDM
#if __has_include(<hip/amd_detail/amd_gfx1250_TDM.h>)
// therock/clang-23 flavor: 6 args (extra 8-dword group before cpol)
#define TDM_LOAD(g0, g1, g2, g3) \
  __builtin_amdgcn_tensor_load_to_lds((g0), (g1), (g2), (g3), (v8i){0,0,0,0,0,0,0,0}, 0)
#else
// ROCm 7.2 / clang-22 flavor: 5 args
#define TDM_LOAD(g0, g1, g2, g3) \
  __builtin_amdgcn_tensor_load_to_lds((g0), (g1), (g2), (g3), 0)
#endif
#endif

// ---------------- small helpers ----------------
__device__ __forceinline__ unsigned short f2bf(float f) {
  unsigned u = __float_as_uint(f);
  u += 0x7fffu + ((u >> 16) & 1u);   // round-to-nearest-even
  return (unsigned short)(u >> 16);
}
__device__ __forceinline__ float gelu_exact(float x) {
  return 0.5f * x * (1.0f + erff(x * 0.70710678118654752440f));
}

// ---------------- generic fp32 -> bf16 cast ----------------
__global__ __launch_bounds__(256)
void k_cast_bf16(const float* __restrict__ src, unsigned short* __restrict__ dst, int n) {
  for (int i = blockIdx.x * 256 + threadIdx.x; i < n; i += gridDim.x * 256)
    dst[i] = f2bf(src[i]);
}

// ---------------- TDM 2D tile load: 128 rows x 32 bf16 -> LDS rows padded to 80B ----------------
#ifdef HAVE_TDM
__device__ __forceinline__ void tdm_load_tile(unsigned long long gaddr, unsigned ldsaddr,
                                              int strideElems)
{
  // D# group0: [1:0]=count=1 | [63:32]=lds_addr | [120:64]=global_addr | [127:126]=type=2
  v4u g0 = { 1u,
             ldsaddr,
             (unsigned)(gaddr & 0xffffffffull),
             (unsigned)((gaddr >> 32) & 0x01ffffffull) | (2u << 30) };
  // D# group1: data_size=1(2B) @17:16 | pad_enable @20 | pad_interval=3(16 dwords) @24:22
  //            pad_amount=3(4 dwords) @31:25 | tensor_dim0=32 @79:48 | tensor_dim1=128 @111:80
  //            tile_dim0=32 @127:112 | tile_dim1=128 @143:128 | dim0_stride=K @207:160
  v8i g1 = { (int)((1u << 16) | (1u << 20) | (3u << 22) | (3u << 25)),
             (int)(32u << 16),    // tensor_dim0[15:0] = 32   (bits 63:48)
             (int)(128u << 16),   // tensor_dim1[15:0] = 128  (bits 95:80)
             (int)(32u << 16),    // tile_dim0 = 32           (bits 127:112)
             (int)(128u),         // tile_dim1 = 128          (bits 143:128)
             (int)strideElems,    // tensor_dim0_stride[31:0] (bits 191:160)
             0, 0 };
  v4i gz = { 0, 0, 0, 0 };
  TDM_LOAD(g0, g1, gz, gz);
}
#endif

// ---------------- WMMA GEMM core: 128x128 tile, bf16 A[*,K] x W[*,K]^T ----------------
struct Acc { v8f t[2][4]; };

template<int K>
__device__ __forceinline__ void gemm_core(const unsigned short* __restrict__ A,
                                          const unsigned short* __restrict__ W,
                                          unsigned char* sm, Acc& acc)
{
  const int tid   = threadIdx.x;
  const int lane  = tid & 31;
  const int wave  = tid >> 5;
  const int wm    = wave & 3;        // 4 row groups of 32
  const int wn    = wave >> 2;       // 2 col groups of 64
  const int lmod  = lane & 15;
  const int lhalf = lane >> 4;
  const int blockRow = blockIdx.x * 128;
  const int blockCol = blockIdx.y * 128;

  // per-lane fragment k-offsets (elements within 32-wide K slab)
  const int aK0 = lhalf ? 8 : 0,  aK1 = lhalf ? 24 : 16;   // A 16x32 layout
  const int bK0 = lhalf ? 8 : 0,  bK1 = lhalf ? 24 : 16;   // B 32x16 layout

  constexpr int NK = K / 32;

#ifdef HAVE_TDM
  // ---- Tensor Data Mover staging (wave 0 drives both tile DMAs; TDM pads rows to 80B) ----
  const bool issuer = (wave == 0);
  const unsigned ldsBase = (unsigned)(size_t)(void*)sm;
  const unsigned long long aBase =
      (unsigned long long)(size_t)(const void*)(A + (size_t)blockRow * K);
  const unsigned long long wBase =
      (unsigned long long)(size_t)(const void*)(W + (size_t)blockCol * K);

  if (issuer) {
    tdm_load_tile(aBase, ldsBase, K);
    tdm_load_tile(wBase, ldsBase + (unsigned)LDS_TILE, K);
    __builtin_amdgcn_s_wait_tensorcnt(0);
  }
  __syncthreads();
#else
  // ---- manual staging fallback: 256 threads move A + B tile per slab (80B LDS rows) ----
  const int srow  = tid >> 1;
  const int soffE = (tid & 1) * 16;
  const int soffB = (tid & 1) * 32;
  const unsigned short* aSrc = A + (size_t)(blockRow + srow) * K + soffE;
  const unsigned short* wSrc = W + (size_t)(blockCol + srow) * K + soffE;
  auto stage = [&](int kk, int buf) {
    const unsigned short* aS = aSrc + (size_t)kk * 32;
    const unsigned short* wS = wSrc + (size_t)kk * 32;
    uint4 a0 = *(const uint4*)(aS + 0);
    uint4 a1 = *(const uint4*)(aS + 8);
    uint4 w0 = *(const uint4*)(wS + 0);
    uint4 w1 = *(const uint4*)(wS + 8);
    unsigned char* d = sm + buf * LDS_STAGE;
    *(uint4*)(d +            srow * LDS_ROW + soffB + 0)  = a0;
    *(uint4*)(d +            srow * LDS_ROW + soffB + 16) = a1;
    *(uint4*)(d + LDS_TILE + srow * LDS_ROW + soffB + 0)  = w0;
    *(uint4*)(d + LDS_TILE + srow * LDS_ROW + soffB + 16) = w1;
  };
  stage(0, 0);
  __syncthreads();
#endif

#pragma unroll 2
  for (int kk = 0; kk < NK; ++kk) {
    const int cur = kk & 1;

#ifdef HAVE_TDM
    if (issuer && (kk + 1 < NK)) {
      const unsigned dst = ldsBase + (unsigned)((cur ^ 1) * LDS_STAGE);
      tdm_load_tile(aBase + (unsigned long long)(kk + 1) * 64u, dst, K);
      tdm_load_tile(wBase + (unsigned long long)(kk + 1) * 64u, dst + (unsigned)LDS_TILE, K);
    }
#else
    if (kk + 1 < NK) stage(kk + 1, cur ^ 1);
#endif
    if (kk + 2 < NK) {   // gfx1250 global_prefetch: one 64B row per thread, 2 slabs ahead
      const int r = tid & 127;
      const unsigned short* p =
          (tid < 128 ? A + (size_t)(blockRow + r) * K
                     : W + (size_t)(blockCol + r) * K) + (size_t)(kk + 2) * 32;
      __builtin_prefetch(p, 0, 1);
    }

    const unsigned char* sa = sm + cur * LDS_STAGE;
    const unsigned char* sb = sa + LDS_TILE;

    union FU { uint4 q[2]; v16bf v; };
    FU af[2], bf[4];
#pragma unroll
    for (int i = 0; i < 2; ++i) {
      const int row = wm * 32 + i * 16 + lmod;
      af[i].q[0] = *(const uint4*)(sa + row * LDS_ROW + aK0 * 2);
      af[i].q[1] = *(const uint4*)(sa + row * LDS_ROW + aK1 * 2);
    }
#pragma unroll
    for (int j = 0; j < 4; ++j) {
      const int col = wn * 64 + j * 16 + lmod;
      bf[j].q[0] = *(const uint4*)(sb + col * LDS_ROW + bK0 * 2);
      bf[j].q[1] = *(const uint4*)(sb + col * LDS_ROW + bK1 * 2);
    }
#pragma unroll
    for (int i = 0; i < 2; ++i)
#pragma unroll
      for (int j = 0; j < 4; ++j)
        acc.t[i][j] = __builtin_amdgcn_wmma_f32_16x16x32_bf16(
            false, af[i].v, false, bf[j].v, (short)0, acc.t[i][j], false, false);

#ifdef HAVE_TDM
    if (issuer && (kk + 1 < NK)) __builtin_amdgcn_s_wait_tensorcnt(0);
#endif
    __syncthreads();
  }
}

// ---------------- GEMM1: h = gelu(x @ W1^T + b1), bf16 out ----------------
__global__ __launch_bounds__(256)
void k_gemm1(const unsigned short* __restrict__ xb,
             const unsigned short* __restrict__ w1b,
             const float* __restrict__ b1,
             unsigned short* __restrict__ hout)
{
  __shared__ __align__(16) unsigned char sm[2 * LDS_STAGE];
  Acc acc;
#pragma unroll
  for (int i = 0; i < 2; ++i)
#pragma unroll
    for (int j = 0; j < 4; ++j) { v8f z = {}; acc.t[i][j] = z; }

  gemm_core<MDIM>(xb, w1b, sm, acc);

  const int lane = threadIdx.x & 31, wave = threadIdx.x >> 5;
  const int wm = wave & 3, wn = wave >> 2, lmod = lane & 15, lhalf = lane >> 4;
  const int blockRow = blockIdx.x * 128, blockCol = blockIdx.y * 128;
#pragma unroll
  for (int i = 0; i < 2; ++i)
#pragma unroll
    for (int j = 0; j < 4; ++j) {
      const int col = blockCol + wn * 64 + j * 16 + lmod;
      const float bias = b1[col];
#pragma unroll
      for (int r = 0; r < 8; ++r) {
        const int row = blockRow + wm * 32 + i * 16 + r + lhalf * 8;
        hout[(size_t)row * CDIM + col] = f2bf(gelu_exact(acc.t[i][j][r] + bias));
      }
    }
}

// ---------------- GEMM2: out = h @ W2^T + b2, optional gate-combine / full store ----------------
__global__ __launch_bounds__(256)
void k_gemm2(const unsigned short* __restrict__ h,
             const unsigned short* __restrict__ w2b,
             const float* __restrict__ b2,
             const float* __restrict__ gate,   // [BN][NEXP] or null
             int gcol,
             float* __restrict__ results,      // accumulate target or null
             int overwrite,
             float* __restrict__ storeFull)    // x1/x2 capture or null
{
  __shared__ __align__(16) unsigned char sm[2 * LDS_STAGE];
  Acc acc;
#pragma unroll
  for (int i = 0; i < 2; ++i)
#pragma unroll
    for (int j = 0; j < 4; ++j) { v8f z = {}; acc.t[i][j] = z; }

  gemm_core<CDIM>(h, w2b, sm, acc);

  const int lane = threadIdx.x & 31, wave = threadIdx.x >> 5;
  const int wm = wave & 3, wn = wave >> 2, lmod = lane & 15, lhalf = lane >> 4;
  const int blockRow = blockIdx.x * 128, blockCol = blockIdx.y * 128;
#pragma unroll
  for (int i = 0; i < 2; ++i)
#pragma unroll
    for (int j = 0; j < 4; ++j) {
      const int col = blockCol + wn * 64 + j * 16 + lmod;
      const float bias = b2[col];
#pragma unroll
      for (int r = 0; r < 8; ++r) {
        const int row = blockRow + wm * 32 + i * 16 + r + lhalf * 8;
        const float v = acc.t[i][j][r] + bias;
        const size_t o = (size_t)row * CDIM + col;
        if (storeFull) storeFull[o] = v;
        if (results) {
          const float add = gate[(size_t)row * NEXP + gcol] * v;
          results[o] = overwrite ? add : (results[o] + add);
        }
      }
    }
}

// ---------------- gate: logits, softmax, top-2, per-block loss partials ----------------
// grid: 72 blocks x 64 threads (9 blocks per batch b; blocks never straddle b)
__global__ __launch_bounds__(64)
void k_gate(const float* __restrict__ x, const float* __restrict__ gw,
            float* __restrict__ w_e, float* __restrict__ partials)
{
  const int t = blockIdx.x * 64 + threadIdx.x;   // token
  float lg[NEXP];
#pragma unroll
  for (int e = 0; e < NEXP; ++e) lg[e] = 0.f;
  const float* xr = x + (size_t)t * MDIM;
  for (int m = 0; m < MDIM; ++m) {
    const float xv = xr[m];
#pragma unroll
    for (int e = 0; e < NEXP; ++e) lg[e] += xv * gw[(size_t)e * MDIM + m];
  }
  float mx = lg[0];
#pragma unroll
  for (int e = 1; e < NEXP; ++e) mx = fmaxf(mx, lg[e]);
  float se = 0.f, sftm[NEXP];
#pragma unroll
  for (int e = 0; e < NEXP; ++e) { sftm[e] = expf(lg[e] - mx); se += sftm[e]; }
  const float lse = mx + logf(se);
  const float inv = 1.f / se;
#pragma unroll
  for (int e = 0; e < NEXP; ++e) sftm[e] *= inv;

  int i1 = 0; float v1 = sftm[0];
#pragma unroll
  for (int e = 1; e < NEXP; ++e) if (sftm[e] > v1) { v1 = sftm[e]; i1 = e; }
  int i2 = -1; float v2 = -1.f;
#pragma unroll
  for (int e = 0; e < NEXP; ++e) if (e != i1 && sftm[e] > v2) { v2 = sftm[e]; i2 = e; }
  const float wsum = v1 + v2;
  const float wa = v1 / wsum, wb = v2 / wsum;
#pragma unroll
  for (int e = 0; e < NEXP; ++e)
    w_e[(size_t)t * NEXP + e] = (e == i1) ? wa : ((e == i2) ? wb : 0.f);

  // block reduce 17 quantities: z^2, softmax[8], top1-onehot[8]
  __shared__ float red[64];
  float q[17];
  q[0] = lse * lse;
#pragma unroll
  for (int e = 0; e < NEXP; ++e) { q[1 + e] = sftm[e]; q[9 + e] = (e == i1) ? 1.f : 0.f; }
  for (int k = 0; k < 17; ++k) {
    red[threadIdx.x] = q[k];
    __syncthreads();
    for (int s = 32; s > 0; s >>= 1) {
      if (threadIdx.x < s) red[threadIdx.x] += red[threadIdx.x + s];
      __syncthreads();
    }
    if (threadIdx.x == 0) partials[blockIdx.x * 17 + k] = red[0];
    __syncthreads();
  }
}

// ---------------- gate reduce: balance_loss, router_z_loss ----------------
__global__ __launch_bounds__(64)
void k_gate_reduce(const float* __restrict__ partials, float* __restrict__ losses)
{
  const int tid = threadIdx.x;
  const int b = tid >> 3, e = tid & 7;
  float proxy = 0.f, dens = 0.f;
  for (int k = 0; k < 9; ++k) {
    const int blk = b * 9 + k;
    proxy += partials[blk * 17 + 1 + e];
    dens  += partials[blk * 17 + 9 + e];
  }
  const float prod = (proxy / (float)NTOK) * (dens / (float)NTOK);
  float zs = 0.f;
  for (int blk = tid; blk < 72; blk += 64) zs += partials[blk * 17];

  __shared__ float red[64];
  red[tid] = prod; __syncthreads();
  for (int s = 32; s > 0; s >>= 1) { if (tid < s) red[tid] += red[tid + s]; __syncthreads(); }
  float balance = red[0]; __syncthreads();       // mean(b,e)*E^2 == plain sum here (B*E == E^2)
  red[tid] = zs; __syncthreads();
  for (int s = 32; s > 0; s >>= 1) { if (tid < s) red[tid] += red[tid + s]; __syncthreads(); }
  if (tid == 0) { losses[0] = balance; losses[1] = red[0] / (float)BN; }
}

// ---------------- Gram partials for diff loss: src [8][GRAMD] fp32 ----------------
// grid: 288 blocks x 256 threads; each block covers 4096 columns
__global__ __launch_bounds__(256)
void k_gram(const float* __restrict__ src, float* __restrict__ partials)
{
  float g[36];
#pragma unroll
  for (int q = 0; q < 36; ++q) g[q] = 0.f;
  for (int it = 0; it < 16; ++it) {
    const int d = blockIdx.x * 4096 + it * 256 + threadIdx.x;
    float v[8];
#pragma unroll
    for (int i = 0; i < 8; ++i) v[i] = src[(size_t)i * GRAMD + d];
    int q = 0;
#pragma unroll
    for (int i = 0; i < 8; ++i)
#pragma unroll
      for (int j = i; j < 8; ++j) g[q++] += v[i] * v[j];
  }
  __shared__ float red[256];
  for (int q = 0; q < 36; ++q) {
    red[threadIdx.x] = g[q];
    __syncthreads();
    for (int s = 128; s > 0; s >>= 1) {
      if (threadIdx.x < s) red[threadIdx.x] += red[threadIdx.x + s];
      __syncthreads();
    }
    if (threadIdx.x == 0) partials[blockIdx.x * 36 + q] = red[0];
    __syncthreads();
  }
}

__device__ __forceinline__ int packed_idx(int i, int j) { // i<=j
  return i * 8 - (i * (i - 1)) / 2 + (j - i);
}

__global__ __launch_bounds__(64)
void k_diff_reduce(const float* __restrict__ p1, const float* __restrict__ p2,
                   float* __restrict__ losses)
{
  __shared__ float G1[36], G2[36];
  const int tid = threadIdx.x;
  if (tid < 36) {
    float s1 = 0.f, s2 = 0.f;
    for (int blk = 0; blk < 288; ++blk) { s1 += p1[blk * 36 + tid]; s2 += p2[blk * 36 + tid]; }
    G1[tid] = s1; G2[tid] = s2;
  }
  __syncthreads();
  if (tid == 0) {
    float na[8], nb[8];
    for (int i = 0; i < 8; ++i) {
      na[i] = sqrtf(G1[packed_idx(i, i)]) + 1e-6f;
      nb[i] = sqrtf(G2[packed_idx(i, i)]) + 1e-6f;
    }
    double tot = 0.0;
    for (int i = 0; i < 8; ++i)
      for (int j = 0; j < 8; ++j) {
        const int a = i < j ? i : j, b = i < j ? j : i;
        const int q = packed_idx(a, b);
        tot += (double)(G1[q] / (na[i] * na[j])) * (double)(G2[q] / (nb[i] * nb[j]));
      }
    const double D2 = (double)GRAMD * (double)GRAMD;
    losses[2] = (float)(tot / D2);
  }
}

// ---------------- launch ----------------
extern "C" void kernel_launch(void* const* d_in, const int* in_sizes, int n_in,
                              void* d_out, int out_size, void* d_ws, size_t ws_size,
                              hipStream_t stream) {
  const float* x      = (const float*)d_in[0];
  const float* gate_w = (const float*)d_in[1];
  const float* p1_w1  = (const float*)d_in[2];
  const float* p1_b1  = (const float*)d_in[3];
  const float* p1_w2  = (const float*)d_in[4];
  const float* p1_b2  = (const float*)d_in[5];
  const float* ew1    = (const float*)d_in[6];
  const float* eb1    = (const float*)d_in[7];
  const float* ew2    = (const float*)d_in[8];
  const float* eb2    = (const float*)d_in[9];
  const float* aw1    = (const float*)d_in[10];
  const float* ab1    = (const float*)d_in[11];
  const float* aw2    = (const float*)d_in[12];
  const float* ab2    = (const float*)d_in[13];

  float* results = (float*)d_out;
  float* losses  = results + (size_t)BN * CDIM;   // balance, z, diff

  // ---- workspace layout (bytes) ----
  char* ws = (char*)d_ws;
  size_t off = 0;
  auto take = [&](size_t bytes) { char* p = ws + off; off = (off + bytes + 255) & ~(size_t)255; return p; };
  unsigned short* xb    = (unsigned short*)take((size_t)BN * MDIM * 2);
  unsigned short* ew1b  = (unsigned short*)take((size_t)NEXP * CDIM * MDIM * 2);
  unsigned short* ew2b  = (unsigned short*)take((size_t)NEXP * CDIM * CDIM * 2);
  unsigned short* aw1b  = (unsigned short*)take((size_t)(NEXP - 1) * CDIM * MDIM * 2);
  unsigned short* aw2b  = (unsigned short*)take((size_t)(NEXP - 1) * CDIM * CDIM * 2);
  unsigned short* p1w1b = (unsigned short*)take((size_t)CDIM * MDIM * 2);
  unsigned short* p1w2b = (unsigned short*)take((size_t)CDIM * CDIM * 2);
  unsigned short* hbuf  = (unsigned short*)take((size_t)BN * CDIM * 2);
  float* x1    = (float*)take((size_t)BN * CDIM * 4);
  float* x2    = (float*)take((size_t)BN * CDIM * 4);
  float* w_e   = (float*)take((size_t)BN * NEXP * 4);
  float* gateP = (float*)take(72 * 17 * 4);
  float* gram1 = (float*)take(288 * 36 * 4);
  float* gram2 = (float*)take(288 * 36 * 4);

  // ---- fp32 -> bf16 conversions ----
  auto cast = [&](const float* s, unsigned short* d, size_t n) {
    int blocks = (int)((n + 255) / 256); if (blocks > 4096) blocks = 4096;
    k_cast_bf16<<<blocks, 256, 0, stream>>>(s, d, (int)n);
  };
  cast(x,     xb,    (size_t)BN * MDIM);
  cast(ew1,   ew1b,  (size_t)NEXP * CDIM * MDIM);
  cast(ew2,   ew2b,  (size_t)NEXP * CDIM * CDIM);
  cast(aw1,   aw1b,  (size_t)(NEXP - 1) * CDIM * MDIM);
  cast(aw2,   aw2b,  (size_t)(NEXP - 1) * CDIM * CDIM);
  cast(p1_w1, p1w1b, (size_t)CDIM * MDIM);
  cast(p1_w2, p1w2b, (size_t)CDIM * CDIM);

  // ---- gating + router losses ----
  k_gate<<<72, 64, 0, stream>>>(x, gate_w, w_e, gateP);
  k_gate_reduce<<<1, 64, 0, stream>>>(gateP, losses);

  const dim3 gg(BN / 128, CDIM / 128);   // 36 x 16
  // ---- experts (sequential launches => deterministic accumulation) ----
  for (int e = 0; e < NEXP; ++e) {
    k_gemm1<<<gg, 256, 0, stream>>>(xb, ew1b + (size_t)e * CDIM * MDIM, eb1 + (size_t)e * CDIM, hbuf);
    k_gemm2<<<gg, 256, 0, stream>>>(hbuf, ew2b + (size_t)e * CDIM * CDIM, eb2 + (size_t)e * CDIM,
                                    w_e, e, results, (e == 0) ? 1 : 0,
                                    (e == 0) ? x2 : nullptr);
    if (e >= 1) {
      k_gemm1<<<gg, 256, 0, stream>>>(xb, aw1b + (size_t)(e - 1) * CDIM * MDIM,
                                      ab1 + (size_t)(e - 1) * CDIM, hbuf);
      k_gemm2<<<gg, 256, 0, stream>>>(hbuf, aw2b + (size_t)(e - 1) * CDIM * CDIM,
                                      ab2 + (size_t)(e - 1) * CDIM,
                                      w_e, e, results, 0, nullptr);
    }
  }
  // ---- p1 probe MLP -> x1 ----
  k_gemm1<<<gg, 256, 0, stream>>>(xb, p1w1b, p1_b1, hbuf);
  k_gemm2<<<gg, 256, 0, stream>>>(hbuf, p1w2b, p1_b2, nullptr, 0, nullptr, 0, x1);

  // ---- diff loss ----
  k_gram<<<288, 256, 0, stream>>>(x1, gram1);
  k_gram<<<288, 256, 0, stream>>>(x2, gram2);
  k_diff_reduce<<<1, 64, 0, stream>>>(gram1, gram2, losses);
}
